// MHABlock_60275571032294
// MI455X (gfx1250) — matrix operand order repaired and checked by reading, not compiled, not run on hardware
//
#include <hip/hip_runtime.h>
#include <hip/hip_bf16.h>
#include <math.h>

typedef __attribute__((ext_vector_type(16))) _Float16 v16h;
typedef __attribute__((ext_vector_type(8)))  _Float16 v8h;
typedef __attribute__((ext_vector_type(8)))  float    v8f;

#define Bb 32
#define Nn 512
#define Ee 128
#define Hh 8
#define KDk 16
#define FFf 512
#define BN (Bb * Nn)          // 16384
#define EPSc 1e-5f
#define NORMc 0.25f           // 1/sqrt(16)

// ---------------------------------------------------------------------------
// WMMA operand loaders (CDNA5 ISA 7.12.2 layouts), all 128-bit vector loads.
// ---------------------------------------------------------------------------

// A: row-major [M x K] f16, 16x32 tile at (row0, k0).
// lane<16: M=lane, elems 0..7 = K 0..7, elems 8..15 = K 16..23
// lane>=16: M=lane-16, elems 0..7 = K 8..15, elems 8..15 = K 24..31
__device__ inline v16h load_A16x32(const _Float16* __restrict__ A, int lda,
                                   int row0, int k0, int lane) {
    int m  = lane & 15;
    int kh = (lane >> 4) * 8;
    const _Float16* p = A + (size_t)(row0 + m) * lda + k0 + kh;
    v8h lo = *(const v8h*)p;          // K kh..kh+7
    v8h hi = *(const v8h*)(p + 16);   // K 16+kh..16+kh+7
    return __builtin_shufflevector(lo, hi, 0,1,2,3,4,5,6,7,8,9,10,11,12,13,14,15);
}

// B: stored COLUMN-major as BT[N x K] f16 (row n = column n of B), 32x16 tile.
// lane<16: N=lane, elems 0..7 = K 0..7, 8..15 = K 16..23 ; lane>=16: N=lane-16, elems 0..7 = K 8..15, 8..15 = K 24..31
__device__ inline v16h load_BT32x16(const _Float16* __restrict__ BT, int ldbt,
                                    int k0, int col0, int lane) {
    int n  = lane & 15;
    int kb = (lane >> 4) * 8;
    const _Float16* p = BT + (size_t)(col0 + n) * ldbt + k0 + kb;
    v8h lo = *(const v8h*)p;
    v8h hi = *(const v8h*)(p + 16);
    return __builtin_shufflevector(lo, hi, 0,1,2,3,4,5,6,7,8,9,10,11,12,13,14,15);
}

// ---------------------------------------------------------------------------
// Elementwise / packing kernels
// ---------------------------------------------------------------------------

__global__ void k_embed(const float* __restrict__ x1, const float* __restrict__ We1,
                        const float* __restrict__ be1,
                        float* __restrict__ hf, _Float16* __restrict__ hh, int total) {
    int idx = blockIdx.x * blockDim.x + threadIdx.x;
    if (idx >= total) return;
    int row = idx / Ee, e = idx % Ee;
    float v = x1[row * 2 + 0] * We1[e] + x1[row * 2 + 1] * We1[Ee + e] + be1[e];
    hf[idx] = v;
    hh[idx] = (_Float16)v;
}

// pack Wq/Wk/Wv [H,E,KD] f32 -> transposed [384 x E] f16 (row c = output column)
__global__ void k_pack_qkvw(const float* __restrict__ Wq, const float* __restrict__ Wk,
                            const float* __restrict__ Wv, _Float16* __restrict__ WT) {
    int idx = blockIdx.x * blockDim.x + threadIdx.x;
    if (idx >= 384 * Ee) return;
    int c = idx / Ee, e = idx % Ee;
    int sec = c >> 7, hc = c & 127;
    int hd = hc >> 4, kd = hc & 15;
    const float* Ws = (sec == 0) ? Wq : (sec == 1) ? Wk : Wv;
    WT[idx] = (_Float16)Ws[(hd * Ee + e) * KDk + kd];
}

// f32 [R x C] -> f16 transposed [C x R]
__global__ void k_transpose_f16(const float* __restrict__ in, _Float16* __restrict__ out,
                                int R, int C) {
    int idx = blockIdx.x * blockDim.x + threadIdx.x;
    if (idx >= R * C) return;
    int r = idx / C, c = idx % C;
    out[(size_t)c * R + r] = (_Float16)in[idx];
}

// qkv f16 [BN x 384] -> qp/kp [hb][N][32] (K-padded rows), vT [hb][16][N]
__global__ void k_pack_attn(const _Float16* __restrict__ qkv,
                            _Float16* __restrict__ qp, _Float16* __restrict__ kp,
                            _Float16* __restrict__ vT) {
    int idx = blockIdx.x * blockDim.x + threadIdx.x;
    if (idx >= Hh * Bb * Nn * 32) return;
    int kd = idx & 31;
    int n  = (idx >> 5) & (Nn - 1);
    int b  = (idx >> 14) & (Bb - 1);
    int hd = idx >> 19;
    int hb = hd * Bb + b;
    size_t row = (size_t)b * Nn + n;
    _Float16 zq = (_Float16)0.0f, zk = (_Float16)0.0f;
    if (kd < KDk) {
        zq = qkv[row * 384 + hd * KDk + kd];
        zk = qkv[row * 384 + 128 + hd * KDk + kd];
        vT[(size_t)hb * KDk * Nn + (size_t)kd * Nn + n] =
            qkv[row * 384 + 256 + hd * KDk + kd];
    }
    qp[((size_t)hb * Nn + n) * 32 + kd] = zq;
    kp[((size_t)hb * Nn + n) * 32 + kd] = zk;
}

// ---------------------------------------------------------------------------
// GEMM: C[MxN] = A[MxK] * B[KxN], B pre-transposed (BT[NxK]).
// 256 thr = 8 waves; each wave computes a 16x64 strip (4 WMMAs per A load).
// ---------------------------------------------------------------------------
__global__ void __launch_bounds__(256)
k_gemm(const _Float16* __restrict__ A, const _Float16* __restrict__ BT,
       const float* __restrict__ bias, const float* __restrict__ res,
       float* __restrict__ Cf, _Float16* __restrict__ Ch,
       int M, int N, int K, int relu) {
    int lane = threadIdx.x & 31;
    int wv   = threadIdx.x >> 5;
    int m0   = blockIdx.y * 128 + wv * 16;
    int n0   = blockIdx.x * 64;
    if (m0 >= M) return;

    v8f c0 = {}, c1 = {}, c2 = {}, c3 = {};
    for (int k0 = 0; k0 < K; k0 += 32) {
        v16h a  = load_A16x32(A, K, m0, k0, lane);
        v16h b0 = load_BT32x16(BT, K, k0, n0,      lane);
        v16h b1 = load_BT32x16(BT, K, k0, n0 + 16, lane);
        v16h b2 = load_BT32x16(BT, K, k0, n0 + 32, lane);
        v16h b3 = load_BT32x16(BT, K, k0, n0 + 48, lane);
        c0 = __builtin_amdgcn_wmma_f32_16x16x32_f16(false, a, false, b0, (short)0, c0, false, false);
        c1 = __builtin_amdgcn_wmma_f32_16x16x32_f16(false, a, false, b1, (short)0, c1, false, false);
        c2 = __builtin_amdgcn_wmma_f32_16x16x32_f16(false, a, false, b2, (short)0, c2, false, false);
        c3 = __builtin_amdgcn_wmma_f32_16x16x32_f16(false, a, false, b3, (short)0, c3, false, false);
    }

    int n  = lane & 15;
    int mh = (lane >> 4) * 8;
    v8f acc[4] = {c0, c1, c2, c3};
#pragma unroll
    for (int t = 0; t < 4; ++t) {
        int col = n0 + t * 16 + n;
        float bv = bias ? bias[col] : 0.0f;
#pragma unroll
        for (int r = 0; r < 8; ++r) {
            size_t row = (size_t)(m0 + mh + r);
            float v = acc[t][r] + bv;
            if (relu) v = fmaxf(v, 0.0f);
            if (res)  v += res[row * N + col];
            if (Cf) Cf[row * N + col] = v;
            if (Ch) Ch[row * N + col] = (_Float16)v;
        }
    }
}

// ---------------------------------------------------------------------------
// Flash attention: grid = (H*B, N/128); 8 waves/block, 16-query tile per wave,
// 32 keys per iteration (2 QK WMMAs + 1 full PV WMMA).
// ---------------------------------------------------------------------------
__global__ void __launch_bounds__(256)
k_attn(const _Float16* __restrict__ qp, const _Float16* __restrict__ kp,
       const _Float16* __restrict__ vT, _Float16* __restrict__ heads) {
    __shared__ _Float16 sP[8][16][32];

    int lane = threadIdx.x & 31;
    int wv   = threadIdx.x >> 5;
    int hb   = blockIdx.x;
    int hd   = hb >> 5;          // / B
    int b    = hb & 31;
    int q0   = blockIdx.y * 128 + wv * 16;

    const _Float16* qslab = qp + (size_t)hb * Nn * 32;
    const _Float16* kslab = kp + (size_t)hb * Nn * 32;
    const _Float16* vslab = vT + (size_t)hb * KDk * Nn;

    // Q tile [16 x 32] (upper K half zero-padded in memory)
    v16h aq = load_A16x32(qslab, 32, q0, 0, lane);

    v8f   acc = {};
    float mrow[8], lrow[8];
#pragma unroll
    for (int r = 0; r < 8; ++r) { mrow[r] = -1e30f; lrow[r] = 0.0f; }

    int n  = lane & 15;
    int mh = (lane >> 4) * 8;

    for (int j = 0; j < Nn / 32; ++j) {
        int key0 = j * 32;

        // S = Q*K^T for 32 keys: kp rows are keys, contiguous K (padded to 32)
        v16h bk0 = load_BT32x16(kslab, 32, 0, key0,      lane);
        v16h bk1 = load_BT32x16(kslab, 32, 0, key0 + 16, lane);
        v8f z = {};
        v8f s0 = __builtin_amdgcn_wmma_f32_16x16x32_f16(false, aq, false, bk0, (short)0, z, false, false);
        v8f s1 = __builtin_amdgcn_wmma_f32_16x16x32_f16(false, aq, false, bk1, (short)0, z, false, false);

        // online softmax per row (rows live in 16-lane groups)
#pragma unroll
        for (int r = 0; r < 8; ++r) {
            float sv0 = s0[r] * NORMc;
            float sv1 = s1[r] * NORMc;
            float tm = fmaxf(sv0, sv1);
#pragma unroll
            for (int off = 1; off < 16; off <<= 1)
                tm = fmaxf(tm, __shfl_xor(tm, off, 32));
            float mn    = fmaxf(mrow[r], tm);
            float scale = __expf(mrow[r] - mn);
            float p0    = __expf(sv0 - mn);
            float p1    = __expf(sv1 - mn);
            float ps    = p0 + p1;
#pragma unroll
            for (int off = 1; off < 16; off <<= 1)
                ps += __shfl_xor(ps, off, 32);
            lrow[r] = lrow[r] * scale + ps;
            acc[r] *= scale;
            mrow[r] = mn;
            sP[wv][mh + r][n]      = (_Float16)p0;   // C-layout -> LDS
            sP[wv][mh + r][16 + n] = (_Float16)p1;
        }
        __syncthreads();

        // P in A-layout (16 x 32)
        v16h ap;
        {
            int m  = lane & 15;
            int kh = (lane >> 4) * 8;
#pragma unroll
            for (int i = 0; i < 8; ++i) ap[i]     = sP[wv][m][kh + i];
#pragma unroll
            for (int i = 0; i < 8; ++i) ap[8 + i] = sP[wv][m][16 + kh + i];
        }
        // V tile as B [32 keys x 16 kd], vT is [kd][N] so reads are contiguous
        v16h bv = load_BT32x16(vslab, Nn, key0, 0, lane);

        acc = __builtin_amdgcn_wmma_f32_16x16x32_f16(false, ap, false, bv, (short)0, acc, false, false);
        __syncthreads();
    }

    // heads[(b*N+q) * 128 + hd*16 + kd]
#pragma unroll
    for (int r = 0; r < 8; ++r) {
        size_t row = (size_t)b * Nn + q0 + mh + r;
        heads[row * Ee + hd * KDk + n] = (_Float16)(acc[r] / lrow[r]);
    }
}

// ---------------------------------------------------------------------------
// BatchNorm over [BN x E], coalesced: each thread owns one channel
// (grid stride 64*256 = 16384 is a multiple of E=128).
// ---------------------------------------------------------------------------
__global__ void k_zerof(float* __restrict__ p, int n) {
    int idx = blockIdx.x * blockDim.x + threadIdx.x;
    if (idx < n) p[idx] = 0.0f;
}

__global__ void k_bn_partial(const float* __restrict__ x,
                             float* __restrict__ gsum, float* __restrict__ gsum2,
                             int total) {
    __shared__ float ssum[Ee], ssum2[Ee];
    if (threadIdx.x < Ee) { ssum[threadIdx.x] = 0.0f; ssum2[threadIdx.x] = 0.0f; }
    __syncthreads();
    int start = blockIdx.x * blockDim.x + threadIdx.x;
    int c = start & (Ee - 1);
    float s = 0.0f, s2 = 0.0f;
    for (int idx = start; idx < total; idx += gridDim.x * blockDim.x) {
        float v = x[idx];
        s += v; s2 += v * v;
    }
    atomicAdd(&ssum[c], s);
    atomicAdd(&ssum2[c], s2);
    __syncthreads();
    if (threadIdx.x < Ee) {
        atomicAdd(&gsum[threadIdx.x],  ssum[threadIdx.x]);
        atomicAdd(&gsum2[threadIdx.x], ssum2[threadIdx.x]);
    }
}

__global__ void k_bn_finalize(const float* __restrict__ gsum, const float* __restrict__ gsum2,
                              float* __restrict__ mean, float* __restrict__ istd) {
    int c = threadIdx.x;
    if (c < Ee) {
        float m   = gsum[c] * (1.0f / BN);
        float var = gsum2[c] * (1.0f / BN) - m * m;
        mean[c] = m;
        istd[c] = rsqrtf(var + EPSc);
    }
}

__global__ void k_bn_apply(const float* __restrict__ x, const float* __restrict__ mean,
                           const float* __restrict__ istd, const float* __restrict__ w,
                           const float* __restrict__ bb,
                           float* __restrict__ of, _Float16* __restrict__ oh, int total) {
    int idx = blockIdx.x * blockDim.x + threadIdx.x;
    if (idx >= total) return;
    int c = idx & (Ee - 1);
    float y = (x[idx] - mean[c]) * istd[c] * w[c] + bb[c];
    if (of) of[idx] = y;
    if (oh) oh[idx] = (_Float16)y;
}

// ---------------------------------------------------------------------------
// Host launcher
// ---------------------------------------------------------------------------
extern "C" void kernel_launch(void* const* d_in, const int* in_sizes, int n_in,
                              void* d_out, int out_size, void* d_ws, size_t ws_size,
                              hipStream_t stream) {
    const float* x1    = (const float*)d_in[0];
    const float* We1   = (const float*)d_in[1];
    const float* be1   = (const float*)d_in[2];
    const float* Wq    = (const float*)d_in[3];
    const float* Wk    = (const float*)d_in[4];
    const float* Wv    = (const float*)d_in[5];
    const float* Wo    = (const float*)d_in[6];
    const float* bn1w  = (const float*)d_in[7];
    const float* bn1b  = (const float*)d_in[8];
    const float* ffW1  = (const float*)d_in[9];
    const float* ffb1  = (const float*)d_in[10];
    const float* ffW2  = (const float*)d_in[11];
    const float* ffb2  = (const float*)d_in[12];
    const float* bn2w  = (const float*)d_in[13];
    const float* bn2b  = (const float*)d_in[14];
    float* out = (float*)d_out;

    char* w = (char*)d_ws;
    auto carve = [&](size_t bytes) {
        void* p = (void*)w;
        w += (bytes + 255) & ~(size_t)255;
        return p;
    };
    float*     hf32  = (float*)    carve((size_t)BN * Ee * 4);
    _Float16*  hf16  = (_Float16*) carve((size_t)BN * Ee * 2);
    _Float16*  WqkvT = (_Float16*) carve((size_t)384 * Ee * 2);
    _Float16*  WoT   = (_Float16*) carve((size_t)Ee * Ee * 2);
    _Float16*  fW1T  = (_Float16*) carve((size_t)FFf * Ee * 2);
    _Float16*  fW2T  = (_Float16*) carve((size_t)Ee * FFf * 2);
    _Float16*  qkv16 = (_Float16*) carve((size_t)BN * 384 * 2);
    _Float16*  qpad  = (_Float16*) carve((size_t)Hh * Bb * Nn * 32 * 2);
    _Float16*  kpad  = (_Float16*) carve((size_t)Hh * Bb * Nn * 32 * 2);
    _Float16*  vT    = (_Float16*) carve((size_t)Hh * Bb * KDk * Nn * 2);
    _Float16*  heads = (_Float16*) carve((size_t)BN * Ee * 2);
    float*     h2    = (float*)    carve((size_t)BN * Ee * 4);
    float*     h3f   = (float*)    carve((size_t)BN * Ee * 4);
    _Float16*  h3h   = (_Float16*) carve((size_t)BN * Ee * 2);
    _Float16*  ff1   = (_Float16*) carve((size_t)BN * FFf * 2);
    float*     x2    = (float*)    carve((size_t)BN * Ee * 4);
    float*     gsum1 = (float*)    carve(Ee * 4);
    float*     gsq1  = (float*)    carve(Ee * 4);
    float*     gsum2 = (float*)    carve(Ee * 4);
    float*     gsq2  = (float*)    carve(Ee * 4);
    float*     mean1 = (float*)    carve(Ee * 4);
    float*     istd1 = (float*)    carve(Ee * 4);
    float*     mean2 = (float*)    carve(Ee * 4);
    float*     istd2 = (float*)    carve(Ee * 4);

    const int T = 256;

    // 1. embedding
    k_embed<<<(BN * Ee + T - 1) / T, T, 0, stream>>>(x1, We1, be1, hf32, hf16, BN * Ee);

    // 2. weight packing (all transposed for contiguous B-column reads)
    k_pack_qkvw<<<(384 * Ee + T - 1) / T, T, 0, stream>>>(Wq, Wk, Wv, WqkvT);
    k_transpose_f16<<<(Ee * Ee + T - 1) / T, T, 0, stream>>>(Wo, WoT, Ee, Ee);    // [128x128] -> [e][hk]
    k_transpose_f16<<<(Ee * FFf + T - 1) / T, T, 0, stream>>>(ffW1, fW1T, Ee, FFf);
    k_transpose_f16<<<(FFf * Ee + T - 1) / T, T, 0, stream>>>(ffW2, fW2T, FFf, Ee);

    // 3. QKV projection: [BN x 128] @ [128 x 384] -> f16
    k_gemm<<<dim3(384 / 64, BN / 128), T, 0, stream>>>(
        hf16, WqkvT, nullptr, nullptr, nullptr, qkv16, BN, 384, Ee, 0);

    // 4. repack for attention
    k_pack_attn<<<(Hh * Bb * Nn * 32 + T - 1) / T, T, 0, stream>>>(qkv16, qpad, kpad, vT);

    // 5. flash attention -> heads f16 [BN x 128]
    k_attn<<<dim3(Hh * Bb, Nn / 128), T, 0, stream>>>(qpad, kpad, vT, heads);

    // 6. output projection + residual
    k_gemm<<<dim3(Ee / 64, BN / 128), T, 0, stream>>>(
        heads, WoT, nullptr, hf32, h2, nullptr, BN, Ee, Ee, 0);

    // 7. BatchNorm 1
    k_zerof<<<1, T, 0, stream>>>(gsum1, Ee * 2);          // gsum1+gsq1 contiguous
    k_bn_partial<<<64, T, 0, stream>>>(h2, gsum1, gsq1, BN * Ee);
    k_bn_finalize<<<1, Ee, 0, stream>>>(gsum1, gsq1, mean1, istd1);
    k_bn_apply<<<(BN * Ee + T - 1) / T, T, 0, stream>>>(
        h2, mean1, istd1, bn1w, bn1b, h3f, h3h, BN * Ee);

    // 8. FFN1: relu(h3 @ ffW1 + ffb1) -> f16
    k_gemm<<<dim3(FFf / 64, BN / 128), T, 0, stream>>>(
        h3h, fW1T, ffb1, nullptr, nullptr, ff1, BN, FFf, Ee, 1);

    // 9. FFN2 + bias + residual
    k_gemm<<<dim3(Ee / 64, BN / 128), T, 0, stream>>>(
        ff1, fW2T, ffb2, h3f, x2, nullptr, BN, Ee, FFf, 0);

    // 10. BatchNorm 2 -> output
    k_zerof<<<1, T, 0, stream>>>(gsum2, Ee * 2);          // gsum2+gsq2 contiguous
    k_bn_partial<<<64, T, 0, stream>>>(x2, gsum2, gsq2, BN * Ee);
    k_bn_finalize<<<1, Ee, 0, stream>>>(gsum2, gsq2, mean2, istd2);
    k_bn_apply<<<(BN * Ee + T - 1) / T, T, 0, stream>>>(
        x2, mean2, istd2, bn2w, bn2b, out, nullptr, BN * Ee);

    (void)in_sizes; (void)n_in; (void)out_size; (void)ws_size;
}
